// EmbeddingTextDecoder_4681514353265
// MI455X (gfx1250) — matrix operand order repaired and checked by reading, not compiled, not run on hardware
//
#include <hip/hip_runtime.h>
#include <hip/hip_bf16.h>
#include <math.h>

// EmbeddingTextDecoder for MI455X (gfx1250, wave32, WMMA + async LDS copies).
// B=64, T=64, V=32000, E=512, H=1024.

typedef _Float16 half_t;
typedef __attribute__((ext_vector_type(16))) _Float16 v16h;
typedef __attribute__((ext_vector_type(8)))  _Float16 v8h;
typedef __attribute__((ext_vector_type(8)))  float    v8f;
typedef __attribute__((ext_vector_type(4)))  int      v4i;

#define NB   64
#define NT   64
#define NV   32000
#define NE   512
#define NH   1024
#define NSTEPS 63          // T-1
#define G4H  (4 * NH)      // 4096

// ---------------------------------------------------------------------------
// WMMA helpers (V_WMMA_F32_16X16X32_F16, wave32)
// ---------------------------------------------------------------------------
__device__ __forceinline__ v8f wmma_16x16x32(v16h a, v16h b, v8f c) {
  // (neg_a, A, neg_b, B, c_mod, C, reuse_a, reuse_b)
  return __builtin_amdgcn_wmma_f32_16x16x32_f16(false, a, false, b, (short)0, c,
                                                false, false);
}

__device__ __forceinline__ v16h make_v16(v8h lo, v8h hi) {
  return __builtin_shufflevector(lo, hi, 0, 1, 2, 3, 4, 5, 6, 7, 8, 9, 10, 11,
                                 12, 13, 14, 15);
}

// A fragment, 16x32 f16 (MxK), rows m0..m0+15, cols k0..k0+31 of a row-major
// matrix with leading dim `ld` (halves). ISA layout: lanes 0-15 -> M=lane,
// halves = K{0..7,16..23}; lanes 16-31 -> K{8..15,24..31}.
__device__ __forceinline__ v16h load_frag_a(const half_t* A, int ld, int m0, int k0) {
  int lane = threadIdx.x & 31;
  int r    = lane & 15;
  int h8   = (lane >> 4) << 3;   // 0 or 8
  const half_t* p = A + (size_t)(m0 + r) * ld + (k0 + h8);
  v8h lo = *(const v8h*)p;          // K = k0+h8 .. +7
  v8h hi = *(const v8h*)(p + 16);   // K = k0+h8+16 .. +7
  return make_v16(lo, hi);
}

// B fragment, 32x16 f16 (KxN) where B[k][n] = W[n0+n][k0+k], W row-major (N x K).
// ISA layout: lane n=lane%16 holds column n; lanes 0-15 K{0..7,16..23}, lanes 16-31 K{8..15,24..31}.
__device__ __forceinline__ v16h load_frag_b(const half_t* W, int ld, int n0, int k0) {
  int lane = threadIdx.x & 31;
  int n    = lane & 15;
  int kh   = (lane >> 4) << 3;   // 0 or 8
  const half_t* p = W + (size_t)(n0 + n) * ld + (k0 + kh);
  v8h lo = *(const v8h*)p;
  v8h hi = *(const v8h*)(p + 16);
  return make_v16(lo, hi);
}

__device__ __forceinline__ float sigmoidf_fast(float x) {
  return 1.0f / (1.0f + __expf(-x));
}

// ---------------------------------------------------------------------------
// CDNA5 async global->LDS copy (GLOBAL_LOAD_ASYNC_TO_LDS_B128, ASYNCcnt)
// ---------------------------------------------------------------------------
__device__ __forceinline__ void async_copy_b128(const half_t* gsrc, half_t* ldst) {
#if __has_builtin(__builtin_amdgcn_global_load_async_to_lds_b128)
  __builtin_amdgcn_global_load_async_to_lds_b128(
      (__attribute__((address_space(1))) v4i*)gsrc,
      (__attribute__((address_space(3))) v4i*)ldst,
      /*offset=*/0, /*cpol=*/0);
#else
  // VDST = LDS byte address (low 32 bits of generic LDS pointer), VADDR = 64b.
  unsigned int l = (unsigned int)(unsigned long long)ldst;
  asm volatile("global_load_async_to_lds_b128 %0, %1, off"
               :
               : "v"(l), "v"(gsrc)
               : "memory");
#endif
}

__device__ __forceinline__ void wait_async_all() {
#if __has_builtin(__builtin_amdgcn_s_wait_asynccnt)
  __builtin_amdgcn_s_wait_asynccnt(0);
#else
  asm volatile("s_wait_asynccnt 0x0" ::: "memory");
#endif
}

// ---------------------------------------------------------------------------
// Prep kernels
// ---------------------------------------------------------------------------
__global__ void cvt_f32_to_f16(const float* __restrict__ src,
                               half_t* __restrict__ dst, int n) {
  for (int i = blockIdx.x * blockDim.x + threadIdx.x; i < n;
       i += gridDim.x * blockDim.x)
    dst[i] = (half_t)src[i];
}

// x_h rows are t*64+b (t-major); x_h[t*64+b][:] = f16(embed_W[captions[b][t]][:])
__global__ void embed_gather(const float* __restrict__ embW,
                             const int* __restrict__ cap,
                             half_t* __restrict__ x_h) {
  int row = blockIdx.x;            // 0 .. 63*64-1
  int t = row >> 6, b = row & 63;
  int idx = cap[b * NT + t];       // captions (B,T), t in 0..62 => [:, :-1]
  const float* src = embW + (size_t)idx * NE;
  half_t* dst = x_h + (size_t)row * NE;
  for (int e = threadIdx.x; e < NE; e += blockDim.x) dst[e] = (half_t)src[e];
}

__global__ void init_state(const float* __restrict__ hidden,
                           const float* __restrict__ cell,
                           half_t* __restrict__ h0_buf0,
                           float* __restrict__ c0,
                           half_t* __restrict__ h1_slot0,
                           float* __restrict__ c1) {
  int i = blockIdx.x * blockDim.x + threadIdx.x;    // 64*1024 elements
  if (i < NB * NH) {
    h0_buf0[i]  = (half_t)hidden[i];
    h1_slot0[i] = (half_t)hidden[NB * NH + i];
    c0[i] = cell[i];
    c1[i] = cell[NB * NH + i];
  }
}

__global__ void zero_t0(float* __restrict__ out) {
  int i = blockIdx.x * blockDim.x + threadIdx.x;    // 64*32000
  if (i < NB * NV) {
    int b = i / NV, v = i - b * NV;
    out[(size_t)b * NT * NV + v] = 0.0f;            // out[b][0][v]
  }
}

// ---------------------------------------------------------------------------
// Software-pipelined 64xK GEMM segment: acc[mt] += A(64xK) @ W(rows nrow..+15)^T
// Prefetches next iteration's fragments before issuing current WMMAs so the
// compiler can overlap VMEM latency with the matrix pipe.
// ---------------------------------------------------------------------------
__device__ __forceinline__ void gemm_seg(const half_t* __restrict__ A, int K,
                                         const half_t* __restrict__ W, int nrow,
                                         v8f acc[4]) {
  v16h bf_n = load_frag_b(W, K, nrow, 0);
  v16h a0_n = load_frag_a(A, K, 0, 0);
  v16h a1_n = load_frag_a(A, K, 16, 0);
  v16h a2_n = load_frag_a(A, K, 32, 0);
  v16h a3_n = load_frag_a(A, K, 48, 0);
  for (int k0 = 0; k0 < K; k0 += 32) {
    v16h bf = bf_n, a0 = a0_n, a1 = a1_n, a2 = a2_n, a3 = a3_n;
    int kn = k0 + 32;
    kn = (kn < K) ? kn : 0;      // wrap: harmless redundant prefetch on last iter
    bf_n = load_frag_b(W, K, nrow, kn);
    a0_n = load_frag_a(A, K, 0, kn);
    a1_n = load_frag_a(A, K, 16, kn);
    a2_n = load_frag_a(A, K, 32, kn);
    a3_n = load_frag_a(A, K, 48, kn);
    acc[0] = wmma_16x16x32(a0, bf, acc[0]);
    acc[1] = wmma_16x16x32(a1, bf, acc[1]);
    acc[2] = wmma_16x16x32(a2, bf, acc[2]);
    acc[3] = wmma_16x16x32(a3, bf, acc[3]);
  }
}

// ---------------------------------------------------------------------------
// Fused LSTM cell: gates = A1 @ W1^T + A2 @ W2^T (+ b_ih + b_hh), then
// activations + state update. One block owns 16 hidden columns across ALL
// four gates (i/f/g/o), so the elementwise part needs no cross-block sync.
// block = 128 threads = 4 waves; wave w computes gate w's 64x16 tile.
// grid = H/16 = 64 blocks.
// ---------------------------------------------------------------------------
__global__ __launch_bounds__(128) void lstm_cell_kernel(
    const half_t* __restrict__ A1, int K1, const half_t* __restrict__ W1,
    const half_t* __restrict__ A2, int K2, const half_t* __restrict__ W2,
    const float* __restrict__ b_ih, const float* __restrict__ b_hh,
    float* __restrict__ c_state, half_t* __restrict__ h_out) {
  __shared__ float lds_g[4 * NB * 16];   // 16 KB: [gate][b][jloc]

  int wave = threadIdx.x >> 5;
  int lane = threadIdx.x & 31;
  int n0   = blockIdx.x * 16;            // column group within H
  int nrow = wave * NH + n0;             // row base in (4H x K) weight

  v8f acc[4] = {v8f{}, v8f{}, v8f{}, v8f{}};
  gemm_seg(A1, K1, W1, nrow, acc);
  gemm_seg(A2, K2, W2, nrow, acc);

  // Spill gate tiles to LDS (C/D layout: VGPR r, lane L -> M = r + (L>=16)*8).
  int h8 = (lane >> 4) << 3;
  int nl = lane & 15;
#pragma unroll
  for (int mt = 0; mt < 4; ++mt) {
#pragma unroll
    for (int r = 0; r < 8; ++r) {
      int m = mt * 16 + r + h8;                     // batch row
      lds_g[wave * (NB * 16) + m * 16 + nl] = acc[mt][r];
    }
  }
  __syncthreads();

  // Elementwise: 64x16 = 1024 elements, 128 threads, 8 each.
  for (int e = threadIdx.x; e < NB * 16; e += 128) {
    int b = e >> 4, j = e & 15;
    int col = n0 + j;                               // column within H
    float gi = lds_g[0 * (NB * 16) + e] + b_ih[col]          + b_hh[col];
    float gf = lds_g[1 * (NB * 16) + e] + b_ih[NH + col]     + b_hh[NH + col];
    float gg = lds_g[2 * (NB * 16) + e] + b_ih[2 * NH + col] + b_hh[2 * NH + col];
    float go = lds_g[3 * (NB * 16) + e] + b_ih[3 * NH + col] + b_hh[3 * NH + col];
    float si = sigmoidf_fast(gi);
    float sf = sigmoidf_fast(gf);
    float so = sigmoidf_fast(go);
    float tg = tanhf(gg);
    size_t sidx = (size_t)b * NH + col;
    float cn = sf * c_state[sidx] + si * tg;
    c_state[sidx] = cn;
    h_out[sidx] = (half_t)(so * tanhf(cn));
  }
}

// ---------------------------------------------------------------------------
// Batched vocab projection: logits(4032 x 32000) = H1(4032 x 1024) @ Wp^T + bp
// scattered into out[b][t+1][v]. grid = (NV/256, 63); block = 256 = 8 waves.
// Wave w computes 64M x 32N (8 accumulators). The 64x1024 f16 A-slice for
// timestep t is staged through double-buffered LDS via async global->LDS
// copies (ASYNCcnt) while WMMAs run on the other buffer. Wp fragments stream
// from global (f16 Wp = 64 MB, L2-resident on MI455X's 192 MB L2) with a
// register prefetch pipeline that runs across chunk boundaries.
// ---------------------------------------------------------------------------
#define KCH  256
#define LDA_C (KCH + 8)   // padded halves: row stride 528B -> banks spread by 4

__global__ __launch_bounds__(256) void logits_kernel(
    const half_t* __restrict__ h1_all,   // 64 slots x (64 x 1024); slot t+1 = step t
    const half_t* __restrict__ Wp,       // (32000 x 1024) f16 row-major
    const float* __restrict__ bp,        // (32000)
    float* __restrict__ out) {           // (64, 64, 32000) f32
  __shared__ half_t a_lds[2][NB * LDA_C];   // 2 x 33792 B (320 KB WGP LDS)

  int t  = blockIdx.y;                   // 0..62
  int n0 = blockIdx.x * 256 + (threadIdx.x >> 5) * 32;
  int lane = threadIdx.x & 31;
  const half_t* A = h1_all + (size_t)(t + 1) * NB * NH;

  // Issue async stage of one 64xKCH chunk into buffer `buf`.
  auto stage = [&](int buf, int kc) {
    for (int i = threadIdx.x; i < NB * KCH / 8; i += 256) {
      int row = i >> 5;                  // 32 x v8h chunks per row
      int cc  = (i & 31) << 3;
      async_copy_b128(A + (size_t)row * NH + kc + cc,
                      &a_lds[buf][row * LDA_C + cc]);
    }
  };

  v8f acc[4][2];
#pragma unroll
  for (int mt = 0; mt < 4; ++mt) {
    acc[mt][0] = v8f{};
    acc[mt][1] = v8f{};
  }

  // Prologue: stage chunk 0, prefetch first B fragments.
  stage(0, 0);
  v16h bf0_n = load_frag_b(Wp, NH, n0,      0);
  v16h bf1_n = load_frag_b(Wp, NH, n0 + 16, 0);
  wait_async_all();
  __syncthreads();

  int cur = 0;
  for (int kc = 0; kc < NH; kc += KCH) {
    if (kc + KCH < NH) stage(cur ^ 1, kc + KCH);   // async fill of next buffer

    for (int k0 = 0; k0 < KCH; k0 += 32) {
      v16h bf0 = bf0_n, bf1 = bf1_n;
      int kg = kc + k0 + 32;                       // global K, crosses chunks
      kg = (kg < NH) ? kg : 0;
      bf0_n = load_frag_b(Wp, NH, n0,      kg);
      bf1_n = load_frag_b(Wp, NH, n0 + 16, kg);
#pragma unroll
      for (int mt = 0; mt < 4; ++mt) {
        v16h af = load_frag_a(&a_lds[cur][0], LDA_C, mt * 16, k0);
        acc[mt][0] = wmma_16x16x32(af, bf0, acc[mt][0]);
        acc[mt][1] = wmma_16x16x32(af, bf1, acc[mt][1]);
      }
    }

    wait_async_all();                              // next buffer resident
    __syncthreads();                               // all waves done with cur
    cur ^= 1;
  }

  // Epilogue: bias + scatter. Row within slot == batch index b.
  int h8 = (lane >> 4) << 3;
  int nl = lane & 15;
#pragma unroll
  for (int mt = 0; mt < 4; ++mt) {
#pragma unroll
    for (int nt = 0; nt < 2; ++nt) {
#pragma unroll
      for (int r = 0; r < 8; ++r) {
        int b = mt * 16 + r + h8;
        int n = n0 + nt * 16 + nl;
        out[((size_t)b * NT + (t + 1)) * NV + n] = acc[mt][nt][r] + bp[n];
      }
    }
  }
}

// ---------------------------------------------------------------------------
// Launch
// ---------------------------------------------------------------------------
extern "C" void kernel_launch(void* const* d_in, const int* in_sizes, int n_in,
                              void* d_out, int out_size, void* d_ws,
                              size_t ws_size, hipStream_t stream) {
  (void)in_sizes; (void)n_in; (void)out_size; (void)ws_size;

  const float* hidden  = (const float*)d_in[0];
  const float* cell    = (const float*)d_in[1];
  const int*   caps    = (const int*)  d_in[2];
  const float* embed_W = (const float*)d_in[3];
  const float* W_ih0   = (const float*)d_in[4];
  const float* W_hh0   = (const float*)d_in[5];
  const float* b_ih0   = (const float*)d_in[6];
  const float* b_hh0   = (const float*)d_in[7];
  const float* W_ih1   = (const float*)d_in[8];
  const float* W_hh1   = (const float*)d_in[9];
  const float* b_ih1   = (const float*)d_in[10];
  const float* b_hh1   = (const float*)d_in[11];
  const float* Wp      = (const float*)d_in[12];
  const float* bp      = (const float*)d_in[13];
  float* out = (float*)d_out;

  // Workspace carve (~109 MB needed)
  char* ws = (char*)d_ws;
  size_t off = 0;
  auto carve = [&](size_t bytes) {
    void* p = ws + off;
    off = (off + bytes + 255) & ~(size_t)255;
    return p;
  };
  half_t* W_ih0_h = (half_t*)carve((size_t)G4H * NE * 2);
  half_t* W_hh0_h = (half_t*)carve((size_t)G4H * NH * 2);
  half_t* W_ih1_h = (half_t*)carve((size_t)G4H * NH * 2);
  half_t* W_hh1_h = (half_t*)carve((size_t)G4H * NH * 2);
  half_t* Wp_h    = (half_t*)carve((size_t)NV * NH * 2);
  half_t* x_h     = (half_t*)carve((size_t)NSTEPS * NB * NE * 2);
  half_t* h0_buf0 = (half_t*)carve((size_t)NB * NH * 2);
  half_t* h0_buf1 = (half_t*)carve((size_t)NB * NH * 2);
  float*  c0      = (float*) carve((size_t)NB * NH * 4);
  float*  c1      = (float*) carve((size_t)NB * NH * 4);
  half_t* h1_all  = (half_t*)carve((size_t)(NSTEPS + 1) * NB * NH * 2);

  // --- prep: f32 -> f16 weight conversion, embedding gather, init state ---
  cvt_f32_to_f16<<<2048, 256, 0, stream>>>(W_ih0, W_ih0_h, G4H * NE);
  cvt_f32_to_f16<<<2048, 256, 0, stream>>>(W_hh0, W_hh0_h, G4H * NH);
  cvt_f32_to_f16<<<2048, 256, 0, stream>>>(W_ih1, W_ih1_h, G4H * NH);
  cvt_f32_to_f16<<<2048, 256, 0, stream>>>(W_hh1, W_hh1_h, G4H * NH);
  cvt_f32_to_f16<<<4096, 256, 0, stream>>>(Wp, Wp_h, NV * NH);
  embed_gather<<<NSTEPS * NB, 256, 0, stream>>>(embed_W, caps, x_h);
  init_state<<<(NB * NH) / 256, 256, 0, stream>>>(hidden, cell, h0_buf0, c0,
                                                  h1_all /*slot 0*/, c1);
  zero_t0<<<(NB * NV + 255) / 256, 256, 0, stream>>>(out);

  // --- recurrence: 63 steps x 2 fused cell kernels ---
  half_t* h0_buf[2] = {h0_buf0, h0_buf1};
  for (int t = 0; t < NSTEPS; ++t) {
    half_t* h0_in  = h0_buf[t & 1];
    half_t* h0_out = h0_buf[(t + 1) & 1];
    half_t* h1_in  = h1_all + (size_t)t * NB * NH;
    half_t* h1_out = h1_all + (size_t)(t + 1) * NB * NH;
    lstm_cell_kernel<<<NH / 16, 128, 0, stream>>>(
        x_h + (size_t)t * NB * NE, NE, W_ih0_h, h0_in, NH, W_hh0_h, b_ih0,
        b_hh0, c0, h0_out);
    lstm_cell_kernel<<<NH / 16, 128, 0, stream>>>(
        h0_out, NH, W_ih1_h, h1_in, NH, W_hh1_h, b_ih1, b_hh1, c1, h1_out);
  }

  // --- batched vocab projection over all 63 timesteps ---
  dim3 grid(NV / 256, NSTEPS);
  logits_kernel<<<grid, 256, 0, stream>>>(h1_all, Wp_h, bp, out);
}
